// LIFRouter_24249385353828
// MI455X (gfx1250) — matrix operand order repaired and checked by reading, not compiled, not run on hardware
//
#include <hip/hip_runtime.h>
#include <hip/hip_bf16.h>

// LIF router for MI455X (gfx1250), wave32 + WMMA.
//   Kernel 1: I = seq @ W^T + b  via V_WMMA_F32_16X16X32_BF16, bf16x3 split
//             (a_hi*b_hi + a_hi*b_lo + a_lo*b_hi) for fp32-class accuracy.
//             Native __bf16 casts -> hardware v_cvt_pk_bf16_f32 (RNE).
//             Each wave owns a 32-row M strip so every B fragment feeds 6 WMMAs.
//   Kernel 2: sequential LIF scan over T + softmax over E.

typedef __attribute__((ext_vector_type(16))) __bf16 v16bf;
typedef __attribute__((ext_vector_type(4)))  __bf16 v4bf;
typedef __attribute__((ext_vector_type(8)))  float  v8f;

#define BETA_   0.9f
#define THRESH_ 1.0f
#define B_ 16
#define T_ 4096
#define D_ 1024
#define E_ 64
#define KC 128   // K-chunk of W staged in LDS per block

// Split f32 into bf16 hi + bf16 residual (RNE both times, hardware cvt).
__device__ inline void split_bf16(float x, __bf16* h, __bf16* l) {
  __bf16 hh = (__bf16)x;
  *h = hh;
  *l = (__bf16)(x - (float)hh);
}

// Build one 16x32 bf16 A fragment pair (hi,lo) in contiguous-k order (same as the B fragments):
// lanes 0-15 hold K {0..15}, lanes 16-31 hold K {16..31}.
__device__ inline void load_a_frag(const float* __restrict__ ap,
                                   v16bf* ah, v16bf* al) {
  float4 x0 = *(const float4*)(ap);
  float4 x1 = *(const float4*)(ap + 4);
  float4 x2 = *(const float4*)(ap + 8);
  float4 x3 = *(const float4*)(ap + 12);
  float f[16] = {x0.x, x0.y, x0.z, x0.w, x1.x, x1.y, x1.z, x1.w,
                 x2.x, x2.y, x2.z, x2.w, x3.x, x3.y, x3.z, x3.w};
  for (int j = 0; j < 16; ++j) {
    __bf16 h, l;
    split_bf16(f[j], &h, &l);
    (*ah)[j] = h;
    (*al)[j] = l;
  }
}

__global__ __launch_bounds__(256)
void lif_gemm(const float* __restrict__ seq, const float* __restrict__ W,
              const float* __restrict__ bias, float* __restrict__ I) {
  __shared__ __bf16 sWhi[E_ * KC];
  __shared__ __bf16 sWlo[E_ * KC];

  const int tid  = threadIdx.x;
  const int lane = tid & 31;
  const int wave = tid >> 5;
  const int half = lane >> 4;   // 0: lanes 0-15, 1: lanes 16-31
  const int l15  = lane & 15;

  // Each wave owns a 32-row (M) strip (two 16-row fragments), all N=64 columns.
  const int rowBase = blockIdx.x * 256 + wave * 32;
  const float* aRow0 = seq + (size_t)(rowBase + l15) * D_;
  const float* aRow1 = aRow0 + (size_t)16 * D_;

  // f32 accumulators, initialized with bias (c[j] = C[j+8*half][l15]).
  v8f acc[2][4];
  for (int nt = 0; nt < 4; ++nt) {
    float bv = bias[nt * 16 + l15];
    for (int j = 0; j < 8; ++j) { acc[0][nt][j] = bv; acc[1][nt][j] = bv; }
  }

  for (int k0 = 0; k0 < D_; k0 += KC) {
    // ---- Stage W[:, k0:k0+KC] into LDS as bf16 hi/lo, layout [e][k] ----
    for (int it = 0; it < 8; ++it) {
      int idx = it * 256 + tid;        // 2048 float4 per chunk
      int e   = idx >> 5;              // 32 float4 per row
      int k4  = (idx & 31) << 2;
      float4 w4 = *(const float4*)(W + (size_t)e * D_ + k0 + k4);
      float wf[4] = {w4.x, w4.y, w4.z, w4.w};
      v4bf h4, l4;
      for (int j = 0; j < 4; ++j) {
        __bf16 h, l;
        split_bf16(wf[j], &h, &l);
        h4[j] = h;
        l4[j] = l;
      }
      *(v4bf*)(sWhi + e * KC + k4) = h4;
      *(v4bf*)(sWlo + e * KC + k4) = l4;
    }
    __syncthreads();

    for (int ks = 0; ks < 4; ++ks) {
      const int c0 = k0 + ks * 32 + 16 * half;  // per-lane K chunk base
      v16bf ah0, al0, ah1, al1;
      load_a_frag(aRow0 + c0, &ah0, &al0);
      load_a_frag(aRow1 + c0, &ah1, &al1);

      for (int nt = 0; nt < 4; ++nt) {
        // B fragment (32x16): lane needs B[k][n]=W[n][k] for 16 contiguous k.
        const __bf16* ph = sWhi + (nt * 16 + l15) * KC + ks * 32 + 16 * half;
        const __bf16* pl = sWlo + (nt * 16 + l15) * KC + ks * 32 + 16 * half;
        v16bf bh = *(const v16bf*)ph;
        v16bf bl = *(const v16bf*)pl;
        acc[0][nt] = __builtin_amdgcn_wmma_f32_16x16x32_bf16(
            false, ah0, false, bh, (short)0, acc[0][nt], false, false);
        acc[0][nt] = __builtin_amdgcn_wmma_f32_16x16x32_bf16(
            false, ah0, false, bl, (short)0, acc[0][nt], false, false);
        acc[0][nt] = __builtin_amdgcn_wmma_f32_16x16x32_bf16(
            false, al0, false, bh, (short)0, acc[0][nt], false, false);
        acc[1][nt] = __builtin_amdgcn_wmma_f32_16x16x32_bf16(
            false, ah1, false, bh, (short)0, acc[1][nt], false, false);
        acc[1][nt] = __builtin_amdgcn_wmma_f32_16x16x32_bf16(
            false, ah1, false, bl, (short)0, acc[1][nt], false, false);
        acc[1][nt] = __builtin_amdgcn_wmma_f32_16x16x32_bf16(
            false, al1, false, bh, (short)0, acc[1][nt], false, false);
      }
    }
    __syncthreads();
  }

  // Store D: VGPR j holds row (j + 8*half), col = nt*16 + l15.
  for (int g = 0; g < 2; ++g)
    for (int nt = 0; nt < 4; ++nt)
      for (int j = 0; j < 8; ++j)
        I[(size_t)(rowBase + g * 16 + j + 8 * half) * E_ + nt * 16 + l15] =
            acc[g][nt][j];
}

__global__ __launch_bounds__(64)
void lif_scan(const float* __restrict__ I, float* __restrict__ out) {
  __shared__ float sU[E_];
  const int b = blockIdx.x;
  const int e = threadIdx.x;
  const float* p = I + (size_t)b * T_ * E_ + e;

  float U = 0.0f;
  for (int t = 0; t < T_; t += 8) {
    float i0 = p[0],      i1 = p[E_],     i2 = p[2 * E_], i3 = p[3 * E_];
    float i4 = p[4 * E_], i5 = p[5 * E_], i6 = p[6 * E_], i7 = p[7 * E_];
    p += 8 * E_;
    // U = beta*U + I_t ; soft reset == min(U, 1)
    U = fminf(fmaf(BETA_, U, i0), THRESH_);
    U = fminf(fmaf(BETA_, U, i1), THRESH_);
    U = fminf(fmaf(BETA_, U, i2), THRESH_);
    U = fminf(fmaf(BETA_, U, i3), THRESH_);
    U = fminf(fmaf(BETA_, U, i4), THRESH_);
    U = fminf(fmaf(BETA_, U, i5), THRESH_);
    U = fminf(fmaf(BETA_, U, i6), THRESH_);
    U = fminf(fmaf(BETA_, U, i7), THRESH_);
  }

  sU[e] = U;
  __syncthreads();
  float mx = -3.402823466e38f;
  for (int j = 0; j < E_; ++j) mx = fmaxf(mx, sU[j]);
  float sum = 0.0f;
  for (int j = 0; j < E_; ++j) sum += expf(sU[j] - mx);
  out[b * E_ + e] = expf(U - mx) / sum;
}

extern "C" void kernel_launch(void* const* d_in, const int* in_sizes, int n_in,
                              void* d_out, int out_size, void* d_ws, size_t ws_size,
                              hipStream_t stream) {
  const float* seq  = (const float*)d_in[0];   // [16,4096,1024] f32
  const float* W    = (const float*)d_in[1];   // [64,1024] f32
  const float* bias = (const float*)d_in[2];   // [64] f32
  float* I = (float*)d_ws;                     // [65536,64] f32 = 16 MiB scratch
  float* out = (float*)d_out;                  // [16,64] f32

  lif_gemm<<<dim3((B_ * T_) / 256), dim3(256), 0, stream>>>(seq, W, bias, I);
  lif_scan<<<dim3(B_), dim3(E_), 0, stream>>>(I, out);
}
